// memristor_dense_45363444580656
// MI455X (gfx1250) — matrix operand order repaired and checked by reading, not compiled, not run on hardware
//
#include <hip/hip_runtime.h>
#include <hip/hip_bf16.h>
#include <cstdint>

// ---------- CDNA5 WMMA types ----------
typedef __attribute__((ext_vector_type(16))) __bf16   v16bf;
typedef __attribute__((ext_vector_type(2)))  __bf16   v2bf;
typedef __attribute__((ext_vector_type(8)))  float    v8f;
typedef __attribute__((ext_vector_type(8)))  uint32_t v8u;

#define V_REF   0.0265f
#define P_EXP   2.8875254f          // log2(7.4)
#define LDSPITCH 40                 // bf16 elems per LDS row (32 + 8 pad) = 80B, 16B aligned

// Native f32->bf16 (RNE) pair conversion; backend picks v_cvt(_pk)_bf16_f32.
__device__ __forceinline__ uint32_t pack2_bf16(float a, float b) {
  v2bf h;
  h.x = (__bf16)a;
  h.y = (__bf16)b;
  return __builtin_bit_cast(uint32_t, h);
}

// Branchless memristor nonlinearity: fV(x) = V_REF * (2*clip(x,0,1))^log2(7.4)
// v=0: v_log_f32 -> -inf, v_exp_f32(-inf) -> 0, no branch needed.
__device__ __forceinline__ float fV_of_x(float x) {
  float v = 2.0f * __builtin_fminf(__builtin_fmaxf(x, 0.0f), 1.0f);
  return V_REF * __builtin_amdgcn_exp2f(P_EXP * __builtin_amdgcn_logf(v));
}

// A fragment (16x32 bf16): lane half lh: K = lh*8..lh*8+7 and lh*8+16..lh*8+23
__device__ __forceinline__ v16bf fragA(const __bf16* row, int lh) {
  const uint4* p = (const uint4*)(row + lh * 8);
  uint4 u0 = p[0];      // K = lh*8 .. +7
  uint4 u1 = p[2];      // K = lh*8+16 .. +23
  v8u v = {u0.x, u0.y, u0.z, u0.w, u1.x, u1.y, u1.z, u1.w};
  return __builtin_bit_cast(v16bf, v);
}
// B fragment (32x16 bf16, stored transposed [n][k]): lane half lh: K = lh*8..lh*8+7 and lh*8+16..lh*8+23
__device__ __forceinline__ v16bf fragB(const __bf16* row, int lh) {
  const uint4* p = (const uint4*)(row + lh * 8);
  uint4 u0 = p[0];
  uint4 u1 = p[2];
  v8u v = {u0.x, u0.y, u0.z, u0.w, u1.x, u1.y, u1.z, u1.w};
  return __builtin_bit_cast(v16bf, v);
}

__device__ __forceinline__ v8f wmma_bf16(v16bf a, v16bf b, v8f c) {
  return __builtin_amdgcn_wmma_f32_16x16x32_bf16(false, a, false, b,
                                                 (short)0, c, false, false);
}

// Split 16 floats into packed bf16 hi/lo planes; 2x uint4 store each to LDS.
__device__ __forceinline__ void split_store(const float* a, __bf16* dh, __bf16* dl) {
  uint32_t hi[8], lo[8];
#pragma unroll
  for (int j = 0; j < 8; ++j) {
    float x0 = a[2 * j], x1 = a[2 * j + 1];
    uint32_t h = pack2_bf16(x0, x1);
    hi[j] = h;
    float h0 = __uint_as_float(h << 16);           // bf16(x0) as f32 (exact)
    float h1 = __uint_as_float(h & 0xFFFF0000u);   // bf16(x1) as f32 (exact)
    lo[j] = pack2_bf16(x0 - h0, x1 - h1);
  }
  uint4 h0v = {hi[0], hi[1], hi[2], hi[3]}, h1v = {hi[4], hi[5], hi[6], hi[7]};
  uint4 l0v = {lo[0], lo[1], lo[2], lo[3]}, l1v = {lo[4], lo[5], lo[6], lo[7]};
  ((uint4*)dh)[0] = h0v; ((uint4*)dh)[1] = h1v;
  ((uint4*)dl)[0] = l0v; ((uint4*)dl)[1] = l1v;
}

// y[b,n] = ( sum_k fV(x[b,k]) * w[k,n] + fV(1)*bias[n] ) / K_V
__global__ __launch_bounds__(256)
void memristor_wmma_gemm(const float* __restrict__ X,   // (batch, K)
                         const float* __restrict__ W,   // (K, N)
                         const float* __restrict__ Bv,  // (N)
                         float* __restrict__ Y,         // (batch, N)
                         int K, int N)
{
  // double-buffered LDS: 8 * 10240B = 80 KB (of 320 KB/WGP)
  __shared__ __align__(16) __bf16 sAhi[2][128 * LDSPITCH];
  __shared__ __align__(16) __bf16 sAlo[2][128 * LDSPITCH];
  __shared__ __align__(16) __bf16 sBhi[2][128 * LDSPITCH];  // transposed: [n][k]
  __shared__ __align__(16) __bf16 sBlo[2][128 * LDSPITCH];

  const int t   = threadIdx.x;
  const int m0  = blockIdx.y * 128;
  const int n0  = blockIdx.x * 128;

  // staging role: 256 threads -> 128 rows/cols x 2 k-halves of 16
  const int sr = t & 127;       // A: row in tile; B: col in tile
  const int sh = t >> 7;        // 0/1: k-half (16 elements)
  const int soff = sr * LDSPITCH + sh * 16;

  // compute role: 8 waves in 4(M) x 2(N)
  const int lane = t & 31;
  const int wave = t >> 5;
  const int wm   = (wave >> 1) * 32;
  const int wn   = (wave & 1) * 64;
  const int lr   = lane & 15;
  const int lh   = lane >> 4;

  v8f acc[2][4];
#pragma unroll
  for (int i = 0; i < 2; ++i)
#pragma unroll
    for (int j = 0; j < 4; ++j) {
      v8f z = {0.f, 0.f, 0.f, 0.f, 0.f, 0.f, 0.f, 0.f};
      acc[i][j] = z;
    }

  const float* gA = X + (size_t)(m0 + sr) * K + sh * 16;

  // ---- raw tile load into registers (coalesced) ----
  auto load_tiles = [&](int k0, float* a, float* b) {
    const float4* pa = (const float4*)(gA + k0);
#pragma unroll
    for (int q = 0; q < 4; ++q) {
      float4 v4 = pa[q];
      a[q * 4 + 0] = v4.x; a[q * 4 + 1] = v4.y;
      a[q * 4 + 2] = v4.z; a[q * 4 + 3] = v4.w;
    }
    const float* pb = W + (size_t)(k0 + sh * 16) * N + n0 + sr;
#pragma unroll
    for (int j = 0; j < 16; ++j) b[j] = pb[(size_t)j * N];
  };

  // ---- fV transform + bf16x2 split + LDS store ----
  auto transform_store = [&](float* a, const float* b, int buf) {
#pragma unroll
    for (int j = 0; j < 16; ++j) a[j] = fV_of_x(a[j]);
    split_store(a, &sAhi[buf][soff], &sAlo[buf][soff]);
    split_store(b, &sBhi[buf][soff], &sBlo[buf][soff]);
  };

  // ---- 24 WMMAs (bf16x3 over 2x4 tiles) from one LDS buffer ----
  auto compute = [&](int buf) {
    v16bf Ah[2], Al[2], Bh[4], Bl[4];
#pragma unroll
    for (int mt = 0; mt < 2; ++mt) {
      const int row = (wm + mt * 16 + lr) * LDSPITCH;
      Ah[mt] = fragA(&sAhi[buf][row], lh);
      Al[mt] = fragA(&sAlo[buf][row], lh);
    }
#pragma unroll
    for (int nt = 0; nt < 4; ++nt) {
      const int row = (wn + nt * 16 + lr) * LDSPITCH;
      Bh[nt] = fragB(&sBhi[buf][row], lh);
      Bl[nt] = fragB(&sBlo[buf][row], lh);
    }
#pragma unroll
    for (int mt = 0; mt < 2; ++mt)
#pragma unroll
      for (int nt = 0; nt < 4; ++nt) {
        acc[mt][nt] = wmma_bf16(Ah[mt], Bh[nt], acc[mt][nt]);
        acc[mt][nt] = wmma_bf16(Ah[mt], Bl[nt], acc[mt][nt]);
        acc[mt][nt] = wmma_bf16(Al[mt], Bh[nt], acc[mt][nt]);
      }
  };

  // ---- prologue: stage tile 0 into buffer 0 ----
  {
    float a[16], b[16];
    load_tiles(0, a, b);
    transform_store(a, b, 0);
  }
  __syncthreads();

  // ---- main loop: compute tile i from buf while staging tile i+1 to buf^1 ----
  int buf = 0;
  for (int k0 = 32; k0 < K; k0 += 32) {
    float a[16], b[16];
    load_tiles(k0, a, b);       // global loads issue early, hidden under WMMAs
    compute(buf);
    transform_store(a, b, buf ^ 1);
    __syncthreads();            // one barrier per K-step
    buf ^= 1;
  }
  compute(buf);                 // last tile

  // ---- epilogue: bias row (ones column * b) and 1/K_V ----
  const float invKV = 1.0f / (2.0f * V_REF);
  const float fOne  = V_REF * 7.4f;     // fV(x=1) = V_REF * 2^log2(7.4)
#pragma unroll
  for (int mt = 0; mt < 2; ++mt)
#pragma unroll
    for (int nt = 0; nt < 4; ++nt) {
      const int gn = n0 + wn + nt * 16 + lr;
      const float bias = fOne * Bv[gn];
      const int gm = m0 + wm + mt * 16 + lh * 8;
      float* o = Y + (size_t)gm * N + gn;
      v8f a = acc[mt][nt];
#pragma unroll
      for (int r = 0; r < 8; ++r)
        o[(size_t)r * N] = (a[r] + bias) * invKV;
    }
}

extern "C" void kernel_launch(void* const* d_in, const int* in_sizes, int n_in,
                              void* d_out, int out_size, void* d_ws, size_t ws_size,
                              hipStream_t stream) {
  (void)n_in; (void)out_size; (void)d_ws; (void)ws_size;
  const float* x = (const float*)d_in[0];
  const float* w = (const float*)d_in[1];
  const float* b = (const float*)d_in[2];
  float* y = (float*)d_out;

  const int n_out = in_sizes[2];                 // 4096
  const int k     = in_sizes[1] / n_out;         // 4096
  const int batch = in_sizes[0] / k;             // 2048

  dim3 grid(n_out / 128, batch / 128);
  memristor_wmma_gemm<<<grid, dim3(256), 0, stream>>>(x, w, b, y, k, n_out);
}
